// MusicAttention_34316788695297
// MI455X (gfx1250) — matrix operand order repaired and checked by reading, not compiled, not run on hardware
//
#include <hip/hip_runtime.h>

// MusicAttention for MI455X (gfx1250, wave32, WMMA f16 16x16x32, f32 accum).
//
// Reference math reproduced exactly, including the gather-then-relative_shift:
//   x[i,c] = q[i] . emb[i-c+2047]            (pre-shift rel, q already /8)
//   rel'[i,j] = x[i, 1023-i+j]      (j <= i)
//             = 0                   (j == i+1)
//             = x[i+1, j-i-2]       (j >= i+2)
//   scores[i,j] = q[i].k[j] + rel'[i,j] + rel_pos_bias[i-j+2047]
//
// All GEMMs via v_wmma_f32_16x16x32_f16. Attention computes S' = K.Q^T so the
// softmax'd P tile in C-layout coincides exactly with the A-operand layout of
// the P@V WMMA (no LDS transpose). V stored transposed [B,H,DH,S] so the P@V
// B-operand is a contiguous per-lane load. Projection GEMM: each wave owns a
// 16x64 tile (4 accumulators share one A load per k-step -> 4:5 wmma:vmem).

typedef _Float16 f16;
typedef __attribute__((ext_vector_type(16))) _Float16 v16h;
typedef __attribute__((ext_vector_type(8)))  float    v8f;
typedef __attribute__((ext_vector_type(4)))  unsigned int u32x4;

#define S_LEN 1024
#define EMB_DIM 1024
#define NHEAD 16
#define DHEAD 64
#define BH 32          // B * H
#define MROWS 2048     // B * S

union V16 { v16h h; u32x4 u[2]; };

static __device__ inline v16h load16(const f16* p) {            // 16 consecutive f16 (16B aligned)
  V16 r; const u32x4* q = (const u32x4*)p; r.u[0] = q[0]; r.u[1] = q[1]; return r.h;
}
// A-operand (16x32 f16): lane kh=lane>>4 selects K-halves {0-15} / {16-31}
static __device__ inline v16h loadA(const f16* prow, int kh) {  // prow -> element (m, kbase)
  V16 r;
  r.u[0] = *(const u32x4*)(prow + 16*kh);
  r.u[1] = *(const u32x4*)(prow + 16*kh + 8);
  return r.h;
}
static __device__ inline v8f vzero() { v8f z = {0.f,0.f,0.f,0.f,0.f,0.f,0.f,0.f}; return z; }
static __device__ inline v8f wmma_(v16h a, v16h b, v8f c) {
  return __builtin_amdgcn_wmma_f32_16x16x32_f16(false, a, false, b, (short)0, c, false, false);
}

// ---------------- elementwise converts ----------------
__global__ __launch_bounds__(256) void cvt_f16_kernel(const float* __restrict__ in,
                                                      f16* __restrict__ out, int n) {
  int i = blockIdx.x * 256 + threadIdx.x;
  if (i < n) out[i] = (f16)in[i];
}

// W[k][n] f32 -> Wt[n][k] f16 (1024x1024), LDS-tiled
__global__ __launch_bounds__(256) void trans_cvt_kernel(const float* __restrict__ in,
                                                        f16* __restrict__ out) {
  __shared__ float t[16][17];
  int kt = blockIdx.x * 16, nt = blockIdx.y * 16;
  t[threadIdx.y][threadIdx.x] = in[(kt + threadIdx.y) * EMB_DIM + nt + threadIdx.x];
  __syncthreads();
  out[(nt + threadIdx.y) * EMB_DIM + kt + threadIdx.x] = (f16)t[threadIdx.x][threadIdx.y];
}

// ---------------- projection GEMM: out = (X[2048x1024] @ Wt^T + b) * scale ----------------
// Each wave computes a 16x64 tile: one A load feeds 4 WMMAs per k-step.
// mode 0: f16 head-split   [bh][s][d]       (Q, K)
// mode 1: f16 head-split T [bh][d][s]       (V)
// mode 2: f32 dense        [m][n]           (final output projection)
__global__ __launch_bounds__(128) void proj_kernel(const f16* __restrict__ X,
                                                   const f16* __restrict__ Wt,
                                                   const float* __restrict__ bias,
                                                   f16* __restrict__ outH,
                                                   float* __restrict__ outF,
                                                   int mode, float scale) {
  int lane = threadIdx.x & 31, wave = threadIdx.x >> 5;
  int qc = lane & 15, kh = lane >> 4;
  int m0 = blockIdx.x * 16;
  int n0 = blockIdx.y * 256 + wave * 64;     // 64 columns per wave (one head slice)
  int m = m0 + qc;                           // A row for this lane
  int n = n0 + qc;                           // base B column for this lane (tile 0)
  const f16* arow  = X + m * EMB_DIM;
  const f16* brow0 = Wt + (n     ) * EMB_DIM;
  const f16* brow1 = Wt + (n + 16) * EMB_DIM;
  const f16* brow2 = Wt + (n + 32) * EMB_DIM;
  const f16* brow3 = Wt + (n + 48) * EMB_DIM;
  v8f acc0 = vzero(), acc1 = vzero(), acc2 = vzero(), acc3 = vzero();
  for (int kb = 0; kb < EMB_DIM; kb += 32) {
    v16h a = loadA(arow + kb, kh);
    v16h b0 = load16(brow0 + kb + 16 * kh);
    v16h b1 = load16(brow1 + kb + 16 * kh);
    v16h b2 = load16(brow2 + kb + 16 * kh);
    v16h b3 = load16(brow3 + kb + 16 * kh);
    acc0 = wmma_(a, b0, acc0);
    acc1 = wmma_(a, b1, acc1);
    acc2 = wmma_(a, b2, acc2);
    acc3 = wmma_(a, b3, acc3);
  }
  int h = n0 >> 6;                           // whole wave tile lies in one head
#pragma unroll
  for (int t = 0; t < 4; ++t) {
    int nt = n + 16 * t;
    int d  = (nt & 63);
    float bn = bias[nt];
    v8f acc = (t == 0) ? acc0 : (t == 1) ? acc1 : (t == 2) ? acc2 : acc3;
#pragma unroll
    for (int r = 0; r < 8; ++r) {
      int mr = m0 + r + 8 * kh;              // C layout: lanes 0-15 M=r, lanes 16-31 M=r+8
      float v = (acc[r] + bn) * scale;
      int bidx = mr >> 10, s = mr & 1023;
      int bh = bidx * NHEAD + h;
      if (mode == 0)      outH[(bh * S_LEN + s) * DHEAD + d] = (f16)v;
      else if (mode == 1) outH[(bh * DHEAD + d) * S_LEN + s] = (f16)v;
      else                outF[mr * EMB_DIM + nt] = v;
    }
  }
}

// ---------------- pre-shift rel tensor: X[i][bh][c] = q[bh,i] . emb[i-c+2047] ----------------
__global__ __launch_bounds__(128) void xbuild_kernel(const f16* __restrict__ Qh,
                                                     const f16* __restrict__ Emb,
                                                     f16* __restrict__ Xrel) {
  int lane = threadIdx.x & 31, wave = threadIdx.x >> 5;
  int qc = lane & 15, kh = lane >> 4;
  int i = blockIdx.x;
  int c = blockIdx.y * 64 + wave * 16 + qc;
  int e = i - c + 2047;                     // in [1024, 3070] always valid
  const f16* brow = Emb + e * DHEAD;
  v8f d0 = vzero(), d1 = vzero();
#pragma unroll
  for (int db = 0; db < DHEAD; db += 32) {
    v16h b  = load16(brow + db + 16 * kh);
    v16h a0 = loadA(Qh + ((qc)      * S_LEN + i) * DHEAD + db, kh);   // bh tile 0..15
    v16h a1 = loadA(Qh + ((16 + qc) * S_LEN + i) * DHEAD + db, kh);   // bh tile 16..31
    d0 = wmma_(a0, b, d0);
    d1 = wmma_(a1, b, d1);
  }
#pragma unroll
  for (int r = 0; r < 8; ++r) {
    int bh = r + 8 * kh;
    Xrel[((i * BH + bh)      << 10) + c] = (f16)d0[r];
    Xrel[((i * BH + bh + 16) << 10) + c] = (f16)d1[r];
  }
}

// shifted rel + bias lookup for score element (row i, col j)
static __device__ inline float relv(const f16* __restrict__ Xrel,
                                    const float* __restrict__ rpb,
                                    int i, int j, int bh) {
  float r = rpb[i - j + 2047];
  if (j == i + 1) return r;
  int row, col;
  if (j <= i) { row = i;     col = 1023 - i + j; }
  else        { row = i + 1; col = j - i - 2;    }
  return r + (float)Xrel[((row * BH + bh) << 10) + col];
}

// ---------------- fused attention (flash-style online softmax) ----------------
// grid (S/64, B*H), block 128 (4 waves); wave owns 16 query rows.
__global__ __launch_bounds__(128) void attn_kernel(const f16* __restrict__ Qh,
                                                   const f16* __restrict__ Kh,
                                                   const f16* __restrict__ VhT,
                                                   const f16* __restrict__ Xrel,
                                                   const float* __restrict__ rpb,
                                                   f16* __restrict__ O16) {
  int lane = threadIdx.x & 31, wave = threadIdx.x >> 5;
  int qc = lane & 15, kh = lane >> 4;
  int bh = blockIdx.y, b = bh >> 4, h = bh & 15;
  int q0 = blockIdx.x * 64 + wave * 16;
  int i = q0 + qc;                                 // this lane's query row

  // B-operand = Q^T (d x q), resident for the whole kernel
  const f16* qrow = Qh + (bh * S_LEN + i) * DHEAD;
  v16h bq0 = load16(qrow + 16 * kh);               // d in [0,32)
  v16h bq1 = load16(qrow + 32 + 16 * kh);          // d in [32,64)

  const f16* kbase = Kh  + bh * S_LEN * DHEAD;
  const f16* vbase = VhT + bh * DHEAD * S_LEN;

  v8f o0 = vzero(), o1 = vzero(), o2 = vzero(), o3 = vzero();
  float mrun = -3.0e38f, lrun = 0.f;
  int srcbase = kh << 3;                           // shuffle source offset for per-row scales

  for (int j0 = 0; j0 < S_LEN; j0 += 32) {
    // S'^T tiles: c0 = rows j0..j0+15, c1 = rows j0+16..j0+31; cols = queries
    v8f c0 = vzero(), c1 = vzero();
    {
      v16h a00 = loadA(kbase + (j0 + qc) * DHEAD, kh);
      v16h a01 = loadA(kbase + (j0 + qc) * DHEAD + 32, kh);
      c0 = wmma_(a00, bq0, c0); c0 = wmma_(a01, bq1, c0);
      v16h a10 = loadA(kbase + (j0 + 16 + qc) * DHEAD, kh);
      v16h a11 = loadA(kbase + (j0 + 16 + qc) * DHEAD + 32, kh);
      c1 = wmma_(a10, bq0, c1); c1 = wmma_(a11, bq1, c1);
    }
    // shifted relative term + positional bias (per-element gather)
#pragma unroll
    for (int v = 0; v < 8; ++v) {
      int j = j0 + 8 * kh + v;
      c0[v] += relv(Xrel, rpb, i, j, bh);
      c1[v] += relv(Xrel, rpb, i, j + 16, bh);
    }
    // online softmax over this lane's 16 j-values + partner lane (lane^16)
    float mt = c0[0];
#pragma unroll
    for (int v = 1; v < 8; ++v) mt = fmaxf(mt, c0[v]);
#pragma unroll
    for (int v = 0; v < 8; ++v) mt = fmaxf(mt, c1[v]);
    mt = fmaxf(mt, __shfl_xor(mt, 16, 32));
    float mnew = fmaxf(mrun, mt);
    float sc = __expf(mrun - mnew);
    float rs = 0.f;
#pragma unroll
    for (int v = 0; v < 8; ++v) { c0[v] = __expf(c0[v] - mnew); rs += c0[v]; }
#pragma unroll
    for (int v = 0; v < 8; ++v) { c1[v] = __expf(c1[v] - mnew); rs += c1[v]; }
    rs += __shfl_xor(rs, 16, 32);
    lrun = lrun * sc + rs;
    mrun = mnew;
    // pack P: C-layout of S'^T == A-operand layout of P (16 q x 32 j)
    v16h pa;
#pragma unroll
    for (int v = 0; v < 8; ++v) { pa[v] = (f16)c0[v]; pa[8 + v] = (f16)c1[v]; }
    // rescale O accumulators (scale per M-row via shuffle broadcast)
#pragma unroll
    for (int r = 0; r < 8; ++r) {
      float s_r = __shfl(sc, r + srcbase, 32);
      o0[r] *= s_r; o1[r] *= s_r; o2[r] *= s_r; o3[r] *= s_r;
    }
    // O += P @ V : B-operand from transposed V, K-halves {0-7,16-23} / {8-15,24-31} per lane
    V16 tv0, tv1, tv2, tv3;
    tv0.u[0] = *(const u32x4*)(vbase + (0 * 16 + qc) * S_LEN + j0 + 8 * kh); tv0.u[1] = *(const u32x4*)(vbase + (0 * 16 + qc) * S_LEN + j0 + 16 + 8 * kh);
    tv1.u[0] = *(const u32x4*)(vbase + (1 * 16 + qc) * S_LEN + j0 + 8 * kh); tv1.u[1] = *(const u32x4*)(vbase + (1 * 16 + qc) * S_LEN + j0 + 16 + 8 * kh);
    tv2.u[0] = *(const u32x4*)(vbase + (2 * 16 + qc) * S_LEN + j0 + 8 * kh); tv2.u[1] = *(const u32x4*)(vbase + (2 * 16 + qc) * S_LEN + j0 + 16 + 8 * kh);
    tv3.u[0] = *(const u32x4*)(vbase + (3 * 16 + qc) * S_LEN + j0 + 8 * kh); tv3.u[1] = *(const u32x4*)(vbase + (3 * 16 + qc) * S_LEN + j0 + 16 + 8 * kh);
    v16h bv0 = tv0.h, bv1 = tv1.h, bv2 = tv2.h, bv3 = tv3.h;
    o0 = wmma_(pa, bv0, o0);
    o1 = wmma_(pa, bv1, o1);
    o2 = wmma_(pa, bv2, o2);
    o3 = wmma_(pa, bv3, o3);
  }
  // normalize and store f16 [m=b*S+s][h*64+d]
  float inv = 1.f / lrun;
  int col = h * DHEAD + qc;
#pragma unroll
  for (int r = 0; r < 8; ++r) {
    float ir = __shfl(inv, r + srcbase, 32);
    int s = q0 + r + 8 * kh;
    f16* dst = O16 + (size_t)(b * S_LEN + s) * EMB_DIM + col;
    dst[0]  = (f16)(o0[r] * ir);
    dst[16] = (f16)(o1[r] * ir);
    dst[32] = (f16)(o2[r] * ir);
    dst[48] = (f16)(o3[r] * ir);
  }
}

// ---------------- launcher ----------------
#define MiB (1024ull * 1024ull)
// workspace map (needs ~101 MiB)
#define OFF_Q16  (0ull)          // query f16            4 MiB
#define OFF_K16  (4ull  * MiB)   // key f16              4 MiB
#define OFF_V16  (8ull  * MiB)   // value f16            4 MiB
#define OFF_WQT  (12ull * MiB)   // Wq^T f16             2 MiB
#define OFF_WKT  (14ull * MiB)
#define OFF_WVT  (16ull * MiB)
#define OFF_WOT  (18ull * MiB)
#define OFF_EMB  (20ull * MiB)   // emb f16            0.5 MiB
#define OFF_QH   (21ull * MiB)   // [bh][s][d] f16       4 MiB
#define OFF_KH   (25ull * MiB)
#define OFF_VHT  (29ull * MiB)   // [bh][d][s] f16       4 MiB
#define OFF_O16  (33ull * MiB)   // attn out f16         4 MiB
#define OFF_X    (37ull * MiB)   // pre-shift rel f16   64 MiB

extern "C" void kernel_launch(void* const* d_in, const int* in_sizes, int n_in,
                              void* d_out, int out_size, void* d_ws, size_t ws_size,
                              hipStream_t stream) {
  const float* query = (const float*)d_in[0];
  const float* key   = (const float*)d_in[1];
  const float* value = (const float*)d_in[2];
  const float* Wq = (const float*)d_in[3];  const float* bq = (const float*)d_in[4];
  const float* Wk = (const float*)d_in[5];  const float* bk = (const float*)d_in[6];
  const float* Wv = (const float*)d_in[7];  const float* bv = (const float*)d_in[8];
  const float* Wo = (const float*)d_in[9];  const float* bo = (const float*)d_in[10];
  const float* emb  = (const float*)d_in[11];
  const float* rpb  = (const float*)d_in[12];

  char* ws = (char*)d_ws;
  f16* q16  = (f16*)(ws + OFF_Q16);
  f16* k16  = (f16*)(ws + OFF_K16);
  f16* v16  = (f16*)(ws + OFF_V16);
  f16* wqt  = (f16*)(ws + OFF_WQT);
  f16* wkt  = (f16*)(ws + OFF_WKT);
  f16* wvt  = (f16*)(ws + OFF_WVT);
  f16* wot  = (f16*)(ws + OFF_WOT);
  f16* emb16= (f16*)(ws + OFF_EMB);
  f16* Qh   = (f16*)(ws + OFF_QH);
  f16* Kh   = (f16*)(ws + OFF_KH);
  f16* VhT  = (f16*)(ws + OFF_VHT);
  f16* O16  = (f16*)(ws + OFF_O16);
  f16* Xrel = (f16*)(ws + OFF_X);

  const int NBSE = MROWS * EMB_DIM;       // 2,097,152
  const int NEMB = 4095 * DHEAD;          //   262,080
  cvt_f16_kernel<<<(NBSE + 255) / 256, 256, 0, stream>>>(query, q16, NBSE);
  cvt_f16_kernel<<<(NBSE + 255) / 256, 256, 0, stream>>>(key,   k16, NBSE);
  cvt_f16_kernel<<<(NBSE + 255) / 256, 256, 0, stream>>>(value, v16, NBSE);
  cvt_f16_kernel<<<(NEMB + 255) / 256, 256, 0, stream>>>(emb, emb16, NEMB);

  dim3 tg(64, 64), tb(16, 16);
  trans_cvt_kernel<<<tg, tb, 0, stream>>>(Wq, wqt);
  trans_cvt_kernel<<<tg, tb, 0, stream>>>(Wk, wkt);
  trans_cvt_kernel<<<tg, tb, 0, stream>>>(Wv, wvt);
  trans_cvt_kernel<<<tg, tb, 0, stream>>>(Wo, wot);

  dim3 pg(MROWS / 16, EMB_DIM / 256);     // (128, 4), 4 waves/block, 16x64 per wave
  proj_kernel<<<pg, 128, 0, stream>>>(q16, wqt, bq, Qh,  nullptr, 0, 0.125f); // /sqrt(DH)
  proj_kernel<<<pg, 128, 0, stream>>>(k16, wkt, bk, Kh,  nullptr, 0, 1.0f);
  proj_kernel<<<pg, 128, 0, stream>>>(v16, wvt, bv, VhT, nullptr, 1, 1.0f);

  xbuild_kernel<<<dim3(S_LEN, S_LEN / 64), 128, 0, stream>>>(Qh, emb16, Xrel);

  attn_kernel<<<dim3(S_LEN / 64, BH), 128, 0, stream>>>(Qh, Kh, VhT, Xrel, rpb, O16);

  proj_kernel<<<pg, 128, 0, stream>>>(O16, wot, bo, nullptr, (float*)d_out, 2, 1.0f);
}
